// BiLSTMCRF_60739427500114
// MI455X (gfx1250) — matrix operand order repaired and checked by reading, not compiled, not run on hardware
//
#include <hip/hip_runtime.h>
#include <cstdint>
#include <cstddef>

// ---------------- problem constants ----------------
#define B_    64
#define S_    512
#define T_    48
#define E_    300
#define EP_   320          // E padded to multiple of 32 (K of bf16 WMMA)
#define H_    256
#define G4_   1024         // 4*H
#define H2_   512          // 2*H
#define NTOK  32768        // S_*B_
#define NEGV  (-10000.0f)

// ---------------- WMMA types & helpers ----------------
typedef __attribute__((ext_vector_type(16))) __bf16 v16bf;
typedef __attribute__((ext_vector_type(8)))  float  v8f;

struct FragBits { uint32_t u[8]; };

static __device__ __forceinline__ v16bf frag_cast(FragBits s) {
  return __builtin_bit_cast(v16bf, s);
}

static __device__ __forceinline__ uint16_t f2bf(float x) {
  uint32_t u = __float_as_uint(x);
  u += 0x7FFFu + ((u >> 16) & 1u);          // round-to-nearest-even
  return (uint16_t)(u >> 16);
}
static __device__ __forceinline__ uint32_t pk2(float lo, float hi) {
  return (uint32_t)f2bf(lo) | ((uint32_t)f2bf(hi) << 16);
}

// A fragment (16-bit A 16x32, ISA 7.12.2): lane holds row M = base + (lane&15).
// elems 0..7  -> K = kb + (lane>=16?8:0) + 0..7   (contiguous, one b128)
// elems 8..15 -> K = kb + 16 + (lane>=16?8:0) + 0..7 (contiguous, one b128)
static __device__ __forceinline__ v16bf load_frag_a(const uint16_t* base, int stride,
                                                    int row, int kb, int lane) {
  const int koff = kb + ((lane & 16) ? 8 : 0);
  const uint16_t* p = base + (size_t)row * stride + koff;
  uint4 lo = *reinterpret_cast<const uint4*>(p);
  uint4 hi = *reinterpret_cast<const uint4*>(p + 16);
  FragBits s{{lo.x, lo.y, lo.z, lo.w, hi.x, hi.y, hi.z, hi.w}};
  return frag_cast(s);
}

// B fragment (16-bit B 32x16): lane holds col N = base + (lane&15); elems 0..7 ->
// K = kb + (lane>=16?8:0) + 0..7, elems 8..15 -> +16.  Source is B^T row-major [N][K].
static __device__ __forceinline__ v16bf load_frag_b(const uint16_t* base, int stride,
                                                    int row, int kb, int lane) {
  const int koff = kb + ((lane & 16) ? 8 : 0);
  const uint16_t* p = base + (size_t)row * stride + koff;
  uint4 lo = *reinterpret_cast<const uint4*>(p);
  uint4 hi = *reinterpret_cast<const uint4*>(p + 16);
  FragBits s{{lo.x, lo.y, lo.z, lo.w, hi.x, hi.y, hi.z, hi.w}};
  return frag_cast(s);
}

static __device__ __forceinline__ v8f wmma_bf16(v16bf a, v16bf b, v8f c) {
  return __builtin_amdgcn_wmma_f32_16x16x32_bf16(false, a, false, b, (short)0, c,
                                                 false, false);
}

static __device__ __forceinline__ float sigm(float x) {
  return 1.0f / (1.0f + __expf(-x));
}

// ---------------- prep kernels ----------------
__global__ void cvt_pad_kernel(const float* __restrict__ src, uint16_t* __restrict__ dst,
                               int N, int K, int KP) {
  int idx = blockIdx.x * blockDim.x + threadIdx.x;
  if (idx >= N * KP) return;
  int n = idx / KP, k = idx - n * KP;
  dst[idx] = (k < K) ? f2bf(src[(size_t)n * K + k]) : (uint16_t)0;
}

__global__ void bias_kernel(const float* __restrict__ a, const float* __restrict__ b,
                            float* __restrict__ o, int n) {
  int i = blockIdx.x * blockDim.x + threadIdx.x;
  if (i < n) o[i] = a[i] + b[i];
}

// Embedding gather + concat -> x_bf16 [NTOK(=s*64+b)][EP_], zero-padded K 300..319
__global__ void embed_kernel(const int* __restrict__ word_ids,
                             const int* __restrict__ flag_ids,
                             const int* __restrict__ pin_ids,
                             const float* __restrict__ ew,
                             const float* __restrict__ ef,
                             const float* __restrict__ ep,
                             uint16_t* __restrict__ xbf) {
  int idx = blockIdx.x * blockDim.x + threadIdx.x;   // < NTOK*EP_
  if (idx >= NTOK * EP_) return;
  int token = idx / EP_, k = idx - token * EP_;
  int s = token >> 6, b = token & 63;
  int src = b * S_ + s;                               // ids are [B][S]
  float v;
  if (k < 200)       v = ew[(size_t)word_ids[src] * 200 + k];
  else if (k < 250)  v = ef[(size_t)flag_ids[src] * 50 + (k - 200)];
  else if (k < 300)  v = ep[(size_t)pin_ids[src] * 50 + (k - 250)];
  else               v = 0.0f;
  xbf[idx] = f2bf(v);
}

// ---------------- big input GEMM: gx = x @ Wih^T + (bih+bhh) ----------------
// one wave = 16x64 output strip; grid.y selects direction
__global__ __launch_bounds__(256) void gemm_gx_kernel(
    const uint16_t* __restrict__ xbf,
    const uint16_t* __restrict__ wih_f, const uint16_t* __restrict__ wih_b,
    const float* __restrict__ bsum_f,   const float* __restrict__ bsum_b,
    float* __restrict__ gxf,            float* __restrict__ gxb) {
  const int dir = blockIdx.y;
  const uint16_t* w  = dir ? wih_b  : wih_f;
  const float*    bs = dir ? bsum_b : bsum_f;
  float*          gx = dir ? gxb    : gxf;

  const int wid  = blockIdx.x * (blockDim.x >> 5) + (threadIdx.x >> 5);
  const int lane = threadIdx.x & 31;
  const int mt   = wid >> 4;          // 0..2047 (16-row tile of tokens)
  const int n4   = wid & 15;          // 0..15  (64-col group)

  v8f acc[4] = {};
  const int arow = mt * 16 + (lane & 15);
  const int brow = (lane & 15);
  for (int ks = 0; ks < EP_ / 32; ++ks) {
    v16bf a = load_frag_a(xbf, EP_, arow, ks * 32, lane);
#pragma unroll
    for (int j = 0; j < 4; ++j) {
      v16bf b = load_frag_b(w, EP_, n4 * 64 + j * 16 + brow, ks * 32, lane);
      acc[j] = wmma_bf16(a, b, acc[j]);
    }
  }
  const int mloc = (lane & 16) ? 8 : 0;
  const int nl   = lane & 15;
#pragma unroll
  for (int j = 0; j < 4; ++j) {
    int col = n4 * 64 + j * 16 + nl;
    float bv = bs[col];
#pragma unroll
    for (int e = 0; e < 8; ++e) {
      int r = mt * 16 + e + mloc;
      gx[(size_t)r * G4_ + col] = acc[j][e] + bv;
    }
  }
}

// ---------------- recurrent LSTM (one workgroup per direction) ----------------
// 32 waves; wave w owns h-col tile ht=w/2 and M tiles {2*(w&1), 2*(w&1)+1};
// cell state c lives in registers, h shared via LDS bf16 (A-matrix of next step).
__global__ __launch_bounds__(1024, 1) void lstm_kernel(
    const float* __restrict__ gxf, const float* __restrict__ gxb,
    const uint16_t* __restrict__ whh_f, const uint16_t* __restrict__ whh_b,
    const float* __restrict__ h0, const float* __restrict__ c0,
    float* __restrict__ hsf, float* __restrict__ hsb) {
  __shared__ uint16_t hbf[B_ * H_];   // 32 KB: current h in bf16

  const int dir = blockIdx.x;
  const float*    gx  = dir ? gxb   : gxf;
  const uint16_t* whh = dir ? whh_b : whh_f;
  float*          hs  = dir ? hsb   : hsf;

  const int tid = threadIdx.x, lane = tid & 31, w = tid >> 5;
  const int ht  = w >> 1;             // 0..15 (16-col tile of H)
  const int mt0 = (w & 1) * 2;        // 0 or 2 (pair of 16-row batch tiles)
  const int nl  = lane & 15;
  const int mloc = (lane & 16) ? 8 : 0;

  // init shared h (bf16) and private c
  for (int i = tid; i < B_ * H_; i += 1024)
    hbf[i] = f2bf(h0[(size_t)dir * B_ * H_ + i]);
  float c[2][8];
#pragma unroll
  for (int mi = 0; mi < 2; ++mi)
#pragma unroll
    for (int e = 0; e < 8; ++e) {
      int m = (mt0 + mi) * 16 + e + mloc;
      c[mi][e] = c0[(size_t)dir * B_ * H_ + (size_t)m * H_ + ht * 16 + nl];
    }
  __syncthreads();

  for (int step = 0; step < S_; ++step) {
    const int s = dir ? (S_ - 1 - step) : step;
    const float* gxs = gx + (size_t)s * B_ * G4_;

    if (step + 1 < S_) {   // global_prefetch next step's gx slice
      const float* nxt = gx + (size_t)(dir ? (S_ - 2 - step) : (step + 1)) * B_ * G4_;
      __builtin_prefetch(nxt + (size_t)tid * 16, 0, 1);
    }

    // accumulator initialized with gx (pre-activation input contribution)
    v8f acc[2][4];
#pragma unroll
    for (int mi = 0; mi < 2; ++mi)
#pragma unroll
      for (int j = 0; j < 4; ++j)
#pragma unroll
        for (int e = 0; e < 8; ++e) {
          int m = (mt0 + mi) * 16 + e + mloc;
          acc[mi][j][e] = gxs[(size_t)m * G4_ + j * H_ + ht * 16 + nl];
        }

    // g += h @ Whh^T  (K = 256 -> 8 bf16 WMMA k-steps)
#pragma unroll
    for (int ks = 0; ks < H_ / 32; ++ks) {
      v16bf a0 = load_frag_a(hbf, H_, mt0 * 16 + (lane & 15), ks * 32, lane);
      v16bf a1 = load_frag_a(hbf, H_, (mt0 + 1) * 16 + (lane & 15), ks * 32, lane);
#pragma unroll
      for (int j = 0; j < 4; ++j) {
        v16bf b = load_frag_b(whh, H_, j * H_ + ht * 16 + (lane & 15), ks * 32, lane);
        acc[0][j] = wmma_bf16(a0, b, acc[0][j]);
        acc[1][j] = wmma_bf16(a1, b, acc[1][j]);
      }
    }
    __syncthreads();   // all reads of hbf done before anyone rewrites it

    // elementwise cell update; gate order i,f,g,o
#pragma unroll
    for (int mi = 0; mi < 2; ++mi)
#pragma unroll
      for (int e = 0; e < 8; ++e) {
        int m   = (mt0 + mi) * 16 + e + mloc;
        int col = ht * 16 + nl;
        float iv = sigm(acc[mi][0][e]);
        float fv = sigm(acc[mi][1][e]);
        float gv = tanhf(acc[mi][2][e]);
        float ov = sigm(acc[mi][3][e]);
        float cn = fv * c[mi][e] + iv * gv;
        c[mi][e] = cn;
        float hv = ov * tanhf(cn);
        hs[((size_t)s * B_ + m) * H_ + col] = hv;
        hbf[m * H_ + col] = f2bf(hv);
      }
    __syncthreads();   // new h visible to everyone for next step
  }
}

// ---------------- output projection: feats = [hs_f|hs_b] @ W_out^T + b ----------------
__global__ __launch_bounds__(256) void proj_kernel(
    const float* __restrict__ hsf, const float* __restrict__ hsb,
    const uint16_t* __restrict__ wout, const float* __restrict__ bout,
    float* __restrict__ feats) {
  const int wid  = blockIdx.x * (blockDim.x >> 5) + (threadIdx.x >> 5); // M tile 0..2047
  const int lane = threadIdx.x & 31;
  const int row  = wid * 16 + (lane & 15);     // token

  v8f acc[3] = {};
  for (int ks = 0; ks < H2_ / 32; ++ks) {
    const float* src = (ks < 8) ? hsf : hsb;   // concat along K (each half 256 = 8 ksteps)
    const int kb   = (ks & 7) * 32;
    const int koff = kb + ((lane & 16) ? 8 : 0);
    const float* p = src + (size_t)row * H_ + koff;
    float4 f0 = *reinterpret_cast<const float4*>(p);
    float4 f1 = *reinterpret_cast<const float4*>(p + 4);
    float4 f2 = *reinterpret_cast<const float4*>(p + 16);
    float4 f3 = *reinterpret_cast<const float4*>(p + 20);
    FragBits s{{pk2(f0.x, f0.y), pk2(f0.z, f0.w), pk2(f1.x, f1.y), pk2(f1.z, f1.w),
                pk2(f2.x, f2.y), pk2(f2.z, f2.w), pk2(f3.x, f3.y), pk2(f3.z, f3.w)}};
    v16bf a = frag_cast(s);
#pragma unroll
    for (int j = 0; j < 3; ++j) {
      v16bf b = load_frag_b(wout, H2_, j * 16 + (lane & 15), ks * 32, lane);
      acc[j] = wmma_bf16(a, b, acc[j]);
    }
  }
  const int mloc = (lane & 16) ? 8 : 0;
  const int nl   = lane & 15;
#pragma unroll
  for (int j = 0; j < 3; ++j) {
    int n = j * 16 + nl;
    float bv = bout[n];
#pragma unroll
    for (int e = 0; e < 8; ++e) {
      int t = wid * 16 + e + mloc;
      feats[(size_t)t * T_ + n] = acc[j][e] + bv;
    }
  }
}

// ---------------- Viterbi forward (one block per batch element) ----------------
__global__ __launch_bounds__(64) void viterbi_kernel(
    const float* __restrict__ feats, const float* __restrict__ trans,
    float* __restrict__ score_out, int* __restrict__ last_tag,
    uint8_t* __restrict__ bp) {
  __shared__ float tr[T_ * T_];
  __shared__ float fv[T_];
  __shared__ float fvn[T_];
  const int b = blockIdx.x, t = threadIdx.x;
  for (int i = t; i < T_ * T_; i += 64) tr[i] = trans[i];
  if (t < T_) fv[t] = NEGV;
  __syncthreads();

  for (int s = 0; s < S_; ++s) {
    if (t < T_) {
      float best = fv[0] + tr[t * T_ + 0];
      int arg = 0;
      for (int p = 1; p < T_; ++p) {
        float v = fv[p] + tr[t * T_ + p];
        if (v > best) { best = v; arg = p; }   // strict > keeps first max (argmax tiebreak)
      }
      fvn[t] = best + feats[((size_t)s * B_ + b) * T_ + t];
      bp[((size_t)s * B_ + b) * T_ + t] = (uint8_t)arg;
    }
    __syncthreads();
    if (t < T_) fv[t] = fvn[t];
    __syncthreads();
  }
  if (t == 0) {
    float best = fv[0];
    int arg = 0;
    for (int p = 1; p < T_; ++p)
      if (fv[p] > best) { best = fv[p]; arg = p; }
    score_out[b] = best;
    last_tag[b] = arg;
  }
}

// ---------------- backtrack ----------------
__global__ __launch_bounds__(64) void backtrack_kernel(
    const int* __restrict__ last_tag, const uint8_t* __restrict__ bp,
    float* __restrict__ out) {
  const int b = threadIdx.x;
  if (b >= B_) return;
  int tag = last_tag[b];
  float* path = out + B_ + (size_t)b * S_;     // best_path [B][S] after path_score[B]
  path[S_ - 1] = (float)tag;
  for (int s = S_ - 1; s >= 1; --s) {
    tag = bp[((size_t)s * B_ + b) * T_ + tag];
    path[s - 1] = (float)tag;
  }
}

// ---------------- host launcher ----------------
extern "C" void kernel_launch(void* const* d_in, const int* in_sizes, int n_in,
                              void* d_out, int out_size, void* d_ws, size_t ws_size,
                              hipStream_t stream) {
  const int*   word_ids = (const int*)  d_in[0];
  const int*   flag_ids = (const int*)  d_in[1];
  const int*   pin_ids  = (const int*)  d_in[2];
  const float* emb_word = (const float*)d_in[3];
  const float* emb_flag = (const float*)d_in[4];
  const float* emb_pin  = (const float*)d_in[5];
  const float* Wih_f    = (const float*)d_in[6];
  const float* Whh_f    = (const float*)d_in[7];
  const float* bih_f    = (const float*)d_in[8];
  const float* bhh_f    = (const float*)d_in[9];
  const float* Wih_b    = (const float*)d_in[10];
  const float* Whh_b    = (const float*)d_in[11];
  const float* bih_b    = (const float*)d_in[12];
  const float* bhh_b    = (const float*)d_in[13];
  const float* h0       = (const float*)d_in[14];
  const float* c0       = (const float*)d_in[15];
  const float* W_out    = (const float*)d_in[16];
  const float* b_out    = (const float*)d_in[17];
  const float* trans    = (const float*)d_in[18];
  float* out = (float*)d_out;

  // workspace carve-out (256B aligned)
  char* wp = (char*)d_ws;
  auto carve = [&](size_t bytes) -> char* {
    char* p = wp;
    wp += (bytes + 255) & ~(size_t)255;
    return p;
  };
  uint16_t* XBF  = (uint16_t*)carve((size_t)NTOK * EP_ * 2);
  uint16_t* WIHF = (uint16_t*)carve((size_t)G4_ * EP_ * 2);
  uint16_t* WIHB = (uint16_t*)carve((size_t)G4_ * EP_ * 2);
  uint16_t* WHHF = (uint16_t*)carve((size_t)G4_ * H_ * 2);
  uint16_t* WHHB = (uint16_t*)carve((size_t)G4_ * H_ * 2);
  uint16_t* WOUT = (uint16_t*)carve((size_t)T_ * H2_ * 2);
  float*    BSF  = (float*)   carve((size_t)G4_ * 4);
  float*    BSB  = (float*)   carve((size_t)G4_ * 4);
  float*    GXF  = (float*)   carve((size_t)NTOK * G4_ * 4);
  float*    GXB  = (float*)   carve((size_t)NTOK * G4_ * 4);
  float*    HSF  = (float*)   carve((size_t)NTOK * H_ * 4);
  float*    HSB  = (float*)   carve((size_t)NTOK * H_ * 4);
  float*    FEATS= (float*)   carve((size_t)NTOK * T_ * 4);
  uint8_t*  BP   = (uint8_t*) carve((size_t)NTOK * T_);
  int*      LAST = (int*)     carve((size_t)B_ * 4);

  // 1) weight conversion to bf16 (+K padding) and bias fusion
  cvt_pad_kernel<<<(G4_ * EP_ + 255) / 256, 256, 0, stream>>>(Wih_f, WIHF, G4_, E_, EP_);
  cvt_pad_kernel<<<(G4_ * EP_ + 255) / 256, 256, 0, stream>>>(Wih_b, WIHB, G4_, E_, EP_);
  cvt_pad_kernel<<<(G4_ * H_  + 255) / 256, 256, 0, stream>>>(Whh_f, WHHF, G4_, H_, H_);
  cvt_pad_kernel<<<(G4_ * H_  + 255) / 256, 256, 0, stream>>>(Whh_b, WHHB, G4_, H_, H_);
  cvt_pad_kernel<<<(T_ * H2_  + 255) / 256, 256, 0, stream>>>(W_out, WOUT, T_, H2_, H2_);
  bias_kernel<<<(G4_ + 255) / 256, 256, 0, stream>>>(bih_f, bhh_f, BSF, G4_);
  bias_kernel<<<(G4_ + 255) / 256, 256, 0, stream>>>(bih_b, bhh_b, BSB, G4_);

  // 2) embedding gather -> x_bf16 [s*64+b][320]
  embed_kernel<<<(NTOK * EP_ + 255) / 256, 256, 0, stream>>>(
      word_ids, flag_ids, pin_ids, emb_word, emb_flag, emb_pin, XBF);

  // 3) input GEMMs for both directions (WMMA bf16)
  {
    dim3 grid(NTOK / (16 * 8) * 16 / 16, 2);   // 4096 blocks x 2 dirs
    grid.x = 4096;
    gemm_gx_kernel<<<grid, 256, 0, stream>>>(XBF, WIHF, WIHB, BSF, BSB, GXF, GXB);
  }

  // 4) sequential recurrence: one WGP-resident workgroup per direction
  lstm_kernel<<<2, 1024, 0, stream>>>(GXF, GXB, WHHF, WHHB, h0, c0, HSF, HSB);

  // 5) output projection (WMMA bf16, N = 48)
  proj_kernel<<<256, 256, 0, stream>>>(HSF, HSB, WOUT, b_out, FEATS);

  // 6) Viterbi forward pass: per-batch blocks, sequential over S
  viterbi_kernel<<<B_, 64, 0, stream>>>(FEATS, trans, out, LAST, BP);

  // 7) backtrack -> d_out = [path_score(64) | best_path(64x512)]
  backtrack_kernel<<<1, 64, 0, stream>>>(LAST, BP, out);
}
